// SpikingMLP_91233695301648
// MI455X (gfx1250) — matrix operand order repaired and checked by reading, not compiled, not run on hardware
//
#include <hip/hip_runtime.h>
#include <hip/hip_bf16.h>

// ---- CDNA5 WMMA types ----
typedef __attribute__((ext_vector_type(16))) __bf16      v16bf;
typedef __attribute__((ext_vector_type(8)))  float       v8f;
typedef __attribute__((ext_vector_type(4)))  unsigned int u32x4;

// Problem dims (from reference): T=4, B=32, N=196, D=768, H=3072
#define T_STEPS 4
#define BN_ROWS 6272   // 32*196
#define D_DIM   768
#define H_DIM   3072

__device__ __forceinline__ unsigned short f32_to_bf16_rne(float f) {
  unsigned int u = __float_as_uint(f);
  unsigned int r = u + 0x7FFFu + ((u >> 16) & 1u);
  return (unsigned short)(r >> 16);
}

__global__ void convert_f32_bf16(const float* __restrict__ in,
                                 unsigned short* __restrict__ out, long n) {
  long i = (long)blockIdx.x * blockDim.x + threadIdx.x;
  if (i < n) out[i] = f32_to_bf16_rne(in[i]);
}

// Fused GEMM (bf16 A,B -> f32 acc) + bias + 4-step LIF epilogue.
//   A  : (4*BNr) x K  bf16 row-major, t-major in rows (row = t*BNr + m)
//   Bm : Ncols x K    bf16 row-major (K contiguous -> B-operand lanes load 32B runs)
//   out: (4*BNr) x Ncols, bf16 spikes (OUT_BF16) or f32 spikes
// Each wave: 16 M-rows x 64 N-cols x all 4 t  => 4x4 v8f accumulators.
template <bool OUT_BF16>
__global__ __launch_bounds__(256)
void gemm4t_lif(const unsigned short* __restrict__ A,
                const unsigned short* __restrict__ Bm,
                const float* __restrict__ bias,
                void* __restrict__ outp,
                int BNr, int Ncols, int K, int tiles_n) {
  const int wave   = blockIdx.x * (blockDim.x >> 5) + (threadIdx.x >> 5);
  const int tile_n = wave % tiles_n;
  const int tile_m = wave / tiles_n;
  const int lane   = threadIdx.x & 31;
  const int l16    = lane & 15;
  const int lhalf  = lane >> 4;   // 0: lanes 0-15, 1: lanes 16-31

  // A-operand ISA layout (16-bit 16x32): lane holds row M=l16; K chunks
  // [lhalf*8, +8) and [16+lhalf*8, +8)  -> two 16B loads per fragment.
  const int rowA = tile_m * 16 + l16;
  size_t aOff[T_STEPS];
#pragma unroll
  for (int t = 0; t < T_STEPS; ++t)
    aOff[t] = ((size_t)t * (size_t)BNr + (size_t)rowA) * (size_t)K + (size_t)(lhalf * 8);

  // B-operand ISA layout (16-bit 32x16): lane = column N=l16; K chunks
  // [lhalf*8, +8) and [16+lhalf*8, +8)  -> two 16B loads per fragment.
  int   colN[4];
  size_t bOff[4];
#pragma unroll
  for (int nn = 0; nn < 4; ++nn) {
    colN[nn] = tile_n * 64 + nn * 16 + l16;
    bOff[nn] = (size_t)colN[nn] * (size_t)K + (size_t)(lhalf * 8);
  }

  v8f acc[T_STEPS][4];
#pragma unroll
  for (int t = 0; t < T_STEPS; ++t)
#pragma unroll
    for (int nn = 0; nn < 4; ++nn)
      acc[t][nn] = (v8f)(0.0f);

#pragma unroll 2
  for (int k0 = 0; k0 < K; k0 += 32) {
    union FB { u32x4 u[2]; v16bf v; } fa[T_STEPS], fb[4];
#pragma unroll
    for (int t = 0; t < T_STEPS; ++t) {
      fa[t].u[0] = *reinterpret_cast<const u32x4*>(A + aOff[t] + k0);
      fa[t].u[1] = *reinterpret_cast<const u32x4*>(A + aOff[t] + k0 + 16);
    }
#pragma unroll
    for (int nn = 0; nn < 4; ++nn) {
      fb[nn].u[0] = *reinterpret_cast<const u32x4*>(Bm + bOff[nn] + k0);
      fb[nn].u[1] = *reinterpret_cast<const u32x4*>(Bm + bOff[nn] + k0 + 16);
    }
#pragma unroll
    for (int t = 0; t < T_STEPS; ++t)
#pragma unroll
      for (int nn = 0; nn < 4; ++nn)
        acc[t][nn] = __builtin_amdgcn_wmma_f32_16x16x32_bf16(
            false, fa[t].v, false, fb[nn].v, (short)0, acc[t][nn], false, false);
  }

  // Epilogue: bias + LIF over t (v=(v+h)/2; spike=v>=1; hard reset) + store.
  // C/D layout: VGPR r -> M = tile_m*16 + r + lhalf*8 ; N = colN[nn].
  float bv[4];
#pragma unroll
  for (int nn = 0; nn < 4; ++nn) bv[nn] = bias[colN[nn]];

  float vmem[4][8];
#pragma unroll
  for (int nn = 0; nn < 4; ++nn)
#pragma unroll
    for (int r = 0; r < 8; ++r) vmem[nn][r] = 0.0f;

  const int mBase = tile_m * 16 + lhalf * 8;
#pragma unroll
  for (int t = 0; t < T_STEPS; ++t) {
#pragma unroll
    for (int nn = 0; nn < 4; ++nn) {
#pragma unroll
      for (int r = 0; r < 8; ++r) {
        float h  = acc[t][nn][r] + bv[nn];
        float vm = 0.5f * (vmem[nn][r] + h);        // decay_input, tau=2, v_reset=0
        bool  sp = (vm >= 1.0f);
        vmem[nn][r] = sp ? 0.0f : vm;               // hard reset
        size_t idx = ((size_t)t * (size_t)BNr + (size_t)(mBase + r)) * (size_t)Ncols
                     + (size_t)colN[nn];
        if (OUT_BF16)
          reinterpret_cast<unsigned short*>(outp)[idx] = sp ? (unsigned short)0x3F80u
                                                           : (unsigned short)0u;
        else
          reinterpret_cast<float*>(outp)[idx] = sp ? 1.0f : 0.0f;
      }
    }
  }
}

extern "C" void kernel_launch(void* const* d_in, const int* in_sizes, int n_in,
                              void* d_out, int out_size, void* d_ws, size_t ws_size,
                              hipStream_t stream) {
  const float* x  = (const float*)d_in[0];  // (4,32,196,768)
  const float* W1 = (const float*)d_in[1];  // (3072,768)
  const float* b1 = (const float*)d_in[2];  // (3072,)
  const float* W2 = (const float*)d_in[3];  // (768,3072)
  const float* b2 = (const float*)d_in[4];  // (768,)
  float* out = (float*)d_out;               // (4,32,196,768) f32 spikes

  const long nx  = (long)T_STEPS * BN_ROWS * D_DIM;   // 19,267,584
  const long nw1 = (long)H_DIM * D_DIM;               // 2,359,296
  const long nw2 = (long)D_DIM * H_DIM;               // 2,359,296
  const long ns1 = (long)T_STEPS * BN_ROWS * H_DIM;   // 77,070,336

  // Workspace layout (bf16 buffers), 256B aligned offsets.
  char* ws = (char*)d_ws;
  size_t off = 0;
  auto take = [&](size_t bytes) {
    char* p = ws + off;
    off = (off + bytes + 255) & ~(size_t)255;
    return p;
  };
  unsigned short* xb  = (unsigned short*)take((size_t)nx  * 2);
  unsigned short* w1b = (unsigned short*)take((size_t)nw1 * 2);
  unsigned short* w2b = (unsigned short*)take((size_t)nw2 * 2);
  unsigned short* s1b = (unsigned short*)take((size_t)ns1 * 2);
  (void)ws_size; (void)in_sizes; (void)n_in; (void)out_size;

  // 1) fp32 -> bf16 conversions
  convert_f32_bf16<<<(int)((nx  + 255) / 256), 256, 0, stream>>>(x,  xb,  nx);
  convert_f32_bf16<<<(int)((nw1 + 255) / 256), 256, 0, stream>>>(W1, w1b, nw1);
  convert_f32_bf16<<<(int)((nw2 + 255) / 256), 256, 0, stream>>>(W2, w2b, nw2);

  // 2) GEMM1 (25088x768 @ 768x3072) fused with LIF1 -> s1 (bf16 spikes)
  {
    const int tiles_m = BN_ROWS / 16;          // 392
    const int tiles_n = H_DIM / 64;            // 48
    const int waves   = tiles_m * tiles_n;     // 18816
    const int blocks  = waves / 8;             // 8 waves per 256-thread block
    gemm4t_lif<true><<<blocks, 256, 0, stream>>>(xb, w1b, b1, (void*)s1b,
                                                 BN_ROWS, H_DIM, D_DIM, tiles_n);
  }

  // 3) GEMM2 (25088x3072 @ 3072x768) fused with LIF2 -> out (f32 spikes)
  {
    const int tiles_m = BN_ROWS / 16;          // 392
    const int tiles_n = D_DIM / 64;            // 12
    const int waves   = tiles_m * tiles_n;     // 4704
    const int blocks  = waves / 8;             // 588
    gemm4t_lif<false><<<blocks, 256, 0, stream>>>(s1b, w2b, b2, (void*)out,
                                                  BN_ROWS, D_DIM, H_DIM, tiles_n);
  }
}
